// HyperNetwork_73332271611973
// MI455X (gfx1250) — matrix operand order repaired and checked by reading, not compiled, not run on hardware
//
#include <hip/hip_runtime.h>

namespace {
constexpr int kF2    = 9;     // F_SIZE*F_SIZE, OUT_CH=1
constexpr int kZDim  = 64;
constexpr int kInCh  = 512;
constexpr int kNz    = 256;   // FEAT*FEAT
constexpr int kBatch = 256;
constexpr int kMRows = 64;          // batch rows per workgroup
constexpr int kALd   = kNz + 8;     // f16 LDS row stride (264*2B = 528B, 16B aligned)
constexpr int kAsLd  = kZDim + 8;   // f32 LDS row stride for stage-1 result
constexpr int kWoLd  = kZDim + 1;   // pad to kill 9-way bank conflict on W_out
}

typedef __attribute__((ext_vector_type(16))) _Float16 v16h;
typedef __attribute__((ext_vector_type(8)))  float    v8f;
typedef __attribute__((ext_vector_type(4)))  int      i4;

struct F16Frag { i4 lo, hi; };

__device__ __forceinline__ v16h load_frag(const _Float16* p0, const _Float16* p1) {
    F16Frag f;
    f.lo = *reinterpret_cast<const i4*>(p0);
    f.hi = *reinterpret_cast<const i4*>(p1);
    return __builtin_bit_cast(v16h, f);
}

__global__ __launch_bounds__(256) void hypernet_wmma_kernel(
    const float* __restrict__ z, const float* __restrict__ W_in,
    const float* __restrict__ b_in, const float* __restrict__ W_out,
    const float* __restrict__ b_out, float* __restrict__ out)
{
    __shared__ __align__(16) _Float16 As[kMRows][kALd];   // x tile, f16
    __shared__ __align__(16) _Float16 Ws[kZDim][kALd];    // W_in[c], f16
    __shared__ float a_s[kMRows][kAsLd];                  // stage-1 result, f32
    __shared__ float Wout_s[kF2][kWoLd];
    __shared__ float bout_s[kF2];

    const int tid  = threadIdx.x;
    const int chan = blockIdx.x & (kInCh - 1);
    const int mblk = blockIdx.x >> 9;        // grid.x = 512 * (256/64) = 2048
    const int row0 = mblk * kMRows;

    // ---- stage tiny shared weights ----
    for (int o = tid; o < kF2 * kZDim; o += 256)
        Wout_s[o / kZDim][o % kZDim] = W_out[o];
    if (tid < kF2) bout_s[tid] = b_out[tid];

    // ---- stage A (x rows) and B (W_in[chan]) into LDS as f16 ----
    const size_t zrs = (size_t)kInCh * kNz;                       // 131072
    const float* zbase = z + (size_t)row0 * zrs + (size_t)chan * kNz;
    for (int i = tid; i < kMRows * (kNz / 4); i += 256) {
        const int r  = i >> 6;               // kNz/4 = 64 float4 per row
        const int c4 = (i & 63) * 4;
        const float4 v = *reinterpret_cast<const float4*>(zbase + (size_t)r * zrs + c4);
        _Float16* d = &As[r][c4];
        d[0] = (_Float16)v.x; d[1] = (_Float16)v.y;
        d[2] = (_Float16)v.z; d[3] = (_Float16)v.w;
    }
    const float* wbase = W_in + (size_t)chan * kZDim * kNz;
    for (int i = tid; i < kZDim * (kNz / 4); i += 256) {
        const int r  = i >> 6;
        const int c4 = (i & 63) * 4;
        const float4 v = *reinterpret_cast<const float4*>(wbase + r * kNz + c4);
        _Float16* d = &Ws[r][c4];
        d[0] = (_Float16)v.x; d[1] = (_Float16)v.y;
        d[2] = (_Float16)v.z; d[3] = (_Float16)v.w;
    }
    __syncthreads();

    // ---- stage 1: per-wave WMMA, a[64x64] = x[64x256] @ W_in[c]^T ----
    const int lane = tid & 31;
    const int wave = tid >> 5;
    const int mt   = wave >> 1;              // 0..3 -> M tile
    const int nt0  = (wave & 1) * 2;         // N tiles nt0, nt0+1

    const int arow  = mt * 16 + (lane & 15);
    const int asel  = (lane >> 4) * 16;      // A: lanes 0-15 K 0..15, lanes 16-31 K 16..31
    const int brow0 = nt0 * 16 + (lane & 15);
    const int bsel  = (lane >> 4) * 16;      // B: lanes 0-15 K 0..15, lanes 16-31 K 16..31

    v8f acc0 = {};
    v8f acc1 = {};
#pragma unroll
    for (int kk = 0; kk < kNz; kk += 32) {
        const v16h a  = load_frag(&As[arow][kk + asel], &As[arow][kk + asel + 8]);
        const v16h b0 = load_frag(&Ws[brow0][kk + bsel], &Ws[brow0][kk + bsel + 8]);
        const v16h b1 = load_frag(&Ws[brow0 + 16][kk + bsel], &Ws[brow0 + 16][kk + bsel + 8]);
        acc0 = __builtin_amdgcn_wmma_f32_16x16x32_f16(false, a, false, b0,
                                                      (short)0, acc0, false, false);
        acc1 = __builtin_amdgcn_wmma_f32_16x16x32_f16(false, a, false, b1,
                                                      (short)0, acc1, false, false);
    }

    // ---- spill a (+ b_in) using the 16x16 f32 C/D layout ----
    const int ncol  = lane & 15;
    const int rbase = mt * 16 + (lane >> 4) * 8;
    const float bin0 = b_in[chan * kZDim + nt0 * 16 + ncol];
    const float bin1 = b_in[chan * kZDim + (nt0 + 1) * 16 + ncol];
#pragma unroll
    for (int j = 0; j < 8; ++j) {
        a_s[rbase + j][nt0 * 16 + ncol]       = acc0[j] + bin0;
        a_s[rbase + j][(nt0 + 1) * 16 + ncol] = acc1[j] + bin1;
    }
    __syncthreads();

    // ---- stage 2: k[b,f] = a[b,:] . W_out[f,:] + b_out[f]; write [B,1,C,3,3] ----
    for (int o = tid; o < kMRows * kF2; o += 256) {
        const int r = o / kF2;
        const int f = o - r * kF2;
        float s = bout_s[f];
#pragma unroll
        for (int zi = 0; zi < kZDim; ++zi)
            s += a_s[r][zi] * Wout_s[f][zi];
        const size_t bg = (size_t)(row0 + r);
        out[bg * (kInCh * kF2) + (size_t)chan * kF2 + f] = s;
    }
}

extern "C" void kernel_launch(void* const* d_in, const int* in_sizes, int n_in,
                              void* d_out, int out_size, void* d_ws, size_t ws_size,
                              hipStream_t stream) {
    (void)in_sizes; (void)n_in; (void)out_size; (void)d_ws; (void)ws_size;
    const float* z     = (const float*)d_in[0];
    const float* W_in  = (const float*)d_in[1];
    const float* b_in  = (const float*)d_in[2];
    const float* W_out = (const float*)d_in[3];
    const float* b_out = (const float*)d_in[4];
    float* out = (float*)d_out;

    dim3 grid(kInCh * (kBatch / kMRows));   // 2048 workgroups
    dim3 block(256);                        // 8 waves (wave32)
    hypernet_wmma_kernel<<<grid, block, 0, stream>>>(z, W_in, b_in, W_out, b_out, out);
}
